// BCA_4406636445956
// MI455X (gfx1250) — matrix operand order repaired and checked by reading, not compiled, not run on hardware
//
#include <hip/hip_runtime.h>
#include <math.h>

typedef __bf16 bf16_t;
typedef __attribute__((ext_vector_type(16))) __bf16 v16bf;
typedef __attribute__((ext_vector_type(8)))  __bf16 v8bf;
typedef __attribute__((ext_vector_type(8)))  float  v8f;
typedef int v4i __attribute__((__vector_size__(4 * sizeof(int))));

// gfx1250 async global->LDS path (ASYNCcnt-tracked), with safe fallback.
#if defined(__AMDGCN__) && __has_builtin(__builtin_amdgcn_global_load_async_to_lds_b128)
#define HAVE_ASYNC_LDS 1
#define GPTR(p) ((__attribute__((address_space(1))) v4i*)(p))
#define LPTR(p) ((__attribute__((address_space(3))) v4i*)(p))
#else
#define HAVE_ASYNC_LDS 0
#endif

__device__ inline void wait_asynccnt0() {
#if HAVE_ASYNC_LDS
#if __has_builtin(__builtin_amdgcn_s_wait_asynccnt)
  __builtin_amdgcn_s_wait_asynccnt(0);
#else
  asm volatile("s_wait_asynccnt 0x0" ::: "memory");
#endif
#endif
}

// ---------------------------------------------------------------------------
// WMMA fragment helpers (gfx1250 wave32 layouts, CDNA5 ISA §7.12.2)
// A 16x32 bf16: lanes0-15 row=lane  K=0..7 (v0-3) & K=16..23 (v4-7)
//               lanes16-31 row=lane-16 K=8..15 & K=24..31
// B 32x16 bf16: lanes0-15 col=lane  K=0..7 & K=16..23 ; lanes16-31 col=lane-16 K=8..15 & K=24..31
// C/D 16x16 f32: vgpr j -> row j (lanes0-15) / j+8 (lanes16-31), col=lane&15
// ---------------------------------------------------------------------------
__device__ inline v16bf lds_load_a_frag(const bf16_t* base, int rs, int lane) {
  const bf16_t* p = base + (lane & 15) * rs + ((lane >> 4) << 3);
  v8bf lo = *(const v8bf*)p;         // K = k0 .. k0+7
  v8bf hi = *(const v8bf*)(p + 16);  // K = k0+16 .. k0+23
  v16bf r;
#pragma unroll
  for (int i = 0; i < 8; ++i) { r[i] = lo[i]; r[i + 8] = hi[i]; }
  return r;
}

__device__ inline v16bf lds_load_b_frag(const bf16_t* base, int rs, int lane) {
  const bf16_t* p = base + (lane & 15) * rs + ((lane >> 4) << 3);
  v8bf lo = *(const v8bf*)p;         // K = k0 .. k0+7
  v8bf hi = *(const v8bf*)(p + 16);  // K = k0+16 .. k0+23
  v16bf r;
#pragma unroll
  for (int i = 0; i < 8; ++i) { r[i] = lo[i]; r[i + 8] = hi[i]; }
  return r;
}

__device__ inline v8f wmma_bf16(v16bf a, v16bf b, v8f c) {
  return __builtin_amdgcn_wmma_f32_16x16x32_bf16(false, a, false, b, (short)0, c,
                                                 false, false);
}

__device__ inline float red16max(float v) {
  v = fmaxf(v, __shfl_xor(v, 1, 32));
  v = fmaxf(v, __shfl_xor(v, 2, 32));
  v = fmaxf(v, __shfl_xor(v, 4, 32));
  v = fmaxf(v, __shfl_xor(v, 8, 32));
  return v;
}
__device__ inline float red16sum(float v) {
  v += __shfl_xor(v, 1, 32);
  v += __shfl_xor(v, 2, 32);
  v += __shfl_xor(v, 4, 32);
  v += __shfl_xor(v, 8, 32);
  return v;
}

// ---------------------------------------------------------------------------
// Utility kernels
// ---------------------------------------------------------------------------
__global__ void k_zero(float* p, int n) {
  int i = blockIdx.x * blockDim.x + threadIdx.x;
  if (i < n) p[i] = 0.0f;
}

__global__ void k_cvt_bf16(const float* s, bf16_t* d, int n) {
  int i = blockIdx.x * blockDim.x + threadIdx.x;
  if (i < n) d[i] = (bf16_t)s[i];
}

// [C][Nn] f32 -> [Nn][C] bf16 per batch (grid.z = batch)
__global__ __launch_bounds__(256)
void k_transpose_cvt(const float* src, bf16_t* dst, int C, int Nn) {
  __shared__ bf16_t tile[32][33];
  long b = blockIdx.z;
  src += b * (long)C * Nn;
  dst += b * (long)C * Nn;
  int n0 = blockIdx.x * 32, c0 = blockIdx.y * 32;
  int tx = threadIdx.x, ty = threadIdx.y;  // 32 x 8
#pragma unroll
  for (int r = 0; r < 4; ++r) {
    int c = c0 + ty + r * 8;
    tile[ty + r * 8][tx] = (bf16_t)src[(long)c * Nn + n0 + tx];
  }
  __syncthreads();
#pragma unroll
  for (int r = 0; r < 4; ++r) {
    int n = n0 + ty + r * 8;
    dst[(long)n * C + c0 + tx] = tile[tx][ty + r * 8];
  }
}

// ---------------------------------------------------------------------------
// Generic GEMM-NT: C[i,j] = sum_k A[i,k] * B[j,k]
// A: bf16 [I,K] row-major, or (AFF) f32 [I,K] with per-k affine a[k]*v + c[k]
// B: bf16 [J,K] row-major.  C: f32 [I,J] row-major.  grid=(I/128, J/64, batch)
// Non-affine path: double-buffered GLOBAL_LOAD_ASYNC_TO_LDS_B128 pipeline.
// ---------------------------------------------------------------------------
template <bool AFF>
__global__ __launch_bounds__(256)
void k_gemm_nt(const void* Av, const float* aA, const float* cA,
               const bf16_t* Bw, float* C, int I, int J, int K,
               long strideA, long strideB, long strideC) {
  __shared__ __align__(16) bf16_t As[2][128][40];
  __shared__ __align__(16) bf16_t Bs[2][64][40];
  int tid = threadIdx.x, lane = tid & 31, wv = tid >> 5;
  int i0 = blockIdx.x * 128, j0 = blockIdx.y * 64;
  long b = blockIdx.z;
  const bf16_t* Ab = (const bf16_t*)Av + (AFF ? 0 : b * strideA);
  const float* Af = (const float*)Av + (AFF ? b * strideA : 0);
  const bf16_t* Bb = Bw + b * strideB;
  float* Cb = C + b * strideC;

  v8f acc[4];
#pragma unroll
  for (int t = 0; t < 4; ++t)
#pragma unroll
    for (int e = 0; e < 8; ++e) acc[t][e] = 0.0f;

  int ar = tid >> 1, ak = (tid & 1) * 16;  // A tile: 128 rows x 32, 16 elems/thr
  int br = tid >> 1, bk = (tid & 1) * 16;  // B tile: 64 rows x 32 (tid<128)

  if (!AFF) {
#if HAVE_ASYNC_LDS
    auto stage_async = [&](int buf, int k0) {
      {
        const bf16_t* g = Ab + (long)(i0 + ar) * K + k0 + ak;
        __builtin_amdgcn_global_load_async_to_lds_b128(GPTR(g),
                                                       LPTR(&As[buf][ar][ak]), 0, 0);
        __builtin_amdgcn_global_load_async_to_lds_b128(GPTR(g + 8),
                                                       LPTR(&As[buf][ar][ak + 8]), 0, 0);
      }
      if (tid < 128) {
        const bf16_t* g = Bb + (long)(j0 + br) * K + k0 + bk;
        __builtin_amdgcn_global_load_async_to_lds_b128(GPTR(g),
                                                       LPTR(&Bs[buf][br][bk]), 0, 0);
        __builtin_amdgcn_global_load_async_to_lds_b128(GPTR(g + 8),
                                                       LPTR(&Bs[buf][br][bk + 8]), 0, 0);
      }
    };
    stage_async(0, 0);  // prologue: tile 0 in flight
    for (int k0 = 0; k0 < K; k0 += 32) {
      int cur = (k0 >> 5) & 1;
      wait_asynccnt0();    // my tile-k LDS writes landed
      __syncthreads();     // everyone's landed; prev compute done with other buf
      if (k0 + 32 < K) stage_async(cur ^ 1, k0 + 32);  // overlap next tile
      v16bf af = lds_load_a_frag(&As[cur][wv * 16][0], 40, lane);
#pragma unroll
      for (int jb = 0; jb < 4; ++jb) {
        v16bf bf = lds_load_b_frag(&Bs[cur][jb * 16][0], 40, lane);
        acc[jb] = wmma_bf16(af, bf, acc[jb]);
      }
      __syncthreads();
    }
#else
    for (int k0 = 0; k0 < K; k0 += 32) {
      __syncthreads();
      {
        const bf16_t* g = Ab + (long)(i0 + ar) * K + k0 + ak;
        *(v8bf*)&As[0][ar][ak] = *(const v8bf*)g;
        *(v8bf*)&As[0][ar][ak + 8] = *(const v8bf*)(g + 8);
      }
      if (tid < 128) {
        const bf16_t* g = Bb + (long)(j0 + br) * K + k0 + bk;
        *(v8bf*)&Bs[0][br][bk] = *(const v8bf*)g;
        *(v8bf*)&Bs[0][br][bk + 8] = *(const v8bf*)(g + 8);
      }
      __syncthreads();
      v16bf af = lds_load_a_frag(&As[0][wv * 16][0], 40, lane);
#pragma unroll
      for (int jb = 0; jb < 4; ++jb) {
        v16bf bf = lds_load_b_frag(&Bs[0][jb * 16][0], 40, lane);
        acc[jb] = wmma_bf16(af, bf, acc[jb]);
      }
    }
#endif
  } else {
    for (int k0 = 0; k0 < K; k0 += 32) {
      __syncthreads();
      {
        const float* g = Af + (long)(i0 + ar) * K + k0 + ak;
#pragma unroll
        for (int e = 0; e < 16; e += 4) {
          float4 v = *(const float4*)(g + e);
          int kk = k0 + ak + e;
          As[0][ar][ak + e + 0] = (bf16_t)(aA[kk + 0] * v.x + cA[kk + 0]);
          As[0][ar][ak + e + 1] = (bf16_t)(aA[kk + 1] * v.y + cA[kk + 1]);
          As[0][ar][ak + e + 2] = (bf16_t)(aA[kk + 2] * v.z + cA[kk + 2]);
          As[0][ar][ak + e + 3] = (bf16_t)(aA[kk + 3] * v.w + cA[kk + 3]);
        }
      }
      if (tid < 128) {
        const bf16_t* g = Bb + (long)(j0 + br) * K + k0 + bk;
        *(v8bf*)&Bs[0][br][bk] = *(const v8bf*)g;
        *(v8bf*)&Bs[0][br][bk + 8] = *(const v8bf*)(g + 8);
      }
      __syncthreads();
      v16bf af = lds_load_a_frag(&As[0][wv * 16][0], 40, lane);
#pragma unroll
      for (int jb = 0; jb < 4; ++jb) {
        v16bf bf = lds_load_b_frag(&Bs[0][jb * 16][0], 40, lane);
        acc[jb] = wmma_bf16(af, bf, acc[jb]);
      }
    }
  }

  int col = lane & 15, half = (lane >> 4) * 8;
#pragma unroll
  for (int jb = 0; jb < 4; ++jb)
#pragma unroll
    for (int j = 0; j < 8; ++j) {
      int row = i0 + wv * 16 + j + half;
      Cb[(long)row * J + j0 + jb * 16 + col] = acc[jb][j];
    }
}

// ---------------------------------------------------------------------------
// Channel stats: channel = fastest dim (64), z layout [rows][64]
// ---------------------------------------------------------------------------
__global__ void k_stats_cols(const float* z, long total, float* sum, float* sq) {
  __shared__ float ssum[64], ssq[64];
  int t = threadIdx.x;
  if (t < 64) { ssum[t] = 0.0f; ssq[t] = 0.0f; }
  __syncthreads();
  float s = 0.0f, q = 0.0f;
  for (long i = (long)blockIdx.x * blockDim.x + t; i < total;
       i += (long)gridDim.x * blockDim.x) {
    float v = z[i];
    s += v; q += v * v;
  }
  atomicAdd(&ssum[t & 63], s);
  atomicAdd(&ssq[t & 63], q);
  __syncthreads();
  if (t < 64) {
    atomicAdd(&sum[t], ssum[t]);
    atomicAdd(&sq[t], ssq[t]);
  }
}

// Channel stats for U [b][ch][rowlen]: one block per (b,ch) row.
__global__ void k_stats_rows(const float* z, int rowlen, int chmask, float* sum,
                             float* sq) {
  __shared__ float ss[8], qq[8];
  long row = blockIdx.x;
  int ch = (int)(row & chmask);
  const float* p = z + row * (long)rowlen;
  float s = 0.0f, q = 0.0f;
  for (int i = threadIdx.x; i < rowlen; i += blockDim.x) {
    float v = p[i];
    s += v; q += v * v;
  }
#pragma unroll
  for (int m = 16; m >= 1; m >>= 1) {
    s += __shfl_xor(s, m, 32);
    q += __shfl_xor(q, m, 32);
  }
  int wv = threadIdx.x >> 5, lane = threadIdx.x & 31;
  if (lane == 0) { ss[wv] = s; qq[wv] = q; }
  __syncthreads();
  if (threadIdx.x == 0) {
    float S = 0.0f, Q = 0.0f;
#pragma unroll
    for (int i = 0; i < 8; ++i) { S += ss[i]; Q += qq[i]; }
    atomicAdd(&sum[ch], S);
    atomicAdd(&sq[ch], Q);
  }
}

__global__ void k_finalize(const float* sum, const float* sq, const float* g,
                           const float* bb, float* a, float* c, int n,
                           float inv_count) {
  int t = blockIdx.x * blockDim.x + threadIdx.x;
  if (t < n) {
    float mean = sum[t] * inv_count;
    float var = sq[t] * inv_count - mean * mean;
    float r = rsqrtf(var + 1e-5f);
    float av = g[t] * r;
    a[t] = av;
    c[t] = bb[t] - mean * av;
  }
}

// ---------------------------------------------------------------------------
// Flash attention: Q=fx [N,64], K=fyproj [N,64], V=fself [N,64], all f32 with
// per-channel BN affine fused into the bf16 conversion. out softmax(QK^T)V.
// Block = 8 waves, 128 query rows; key tiles of 64. grid = (N/128, B).
// ---------------------------------------------------------------------------
__global__ __launch_bounds__(256)
void k_attn(const float* Qsrc, const float* aq, const float* cq,
            const float* Ksrc, const float* ak, const float* ck,
            const float* Vsrc, const float* av, const float* cv, bf16_t* Fout,
            int Nn) {
  __shared__ __align__(16) bf16_t Qs[128][72];  // Q stage, reused as P stage
  __shared__ __align__(16) bf16_t Kt[64][72];   // Kt[key][m]
  __shared__ __align__(16) bf16_t Vt[64][72];   // Vt[m][key] (transposed)
  int tid = threadIdx.x, lane = tid & 31, wv = tid >> 5;
  int q0 = blockIdx.x * 128;
  long boff = (long)blockIdx.y * Nn * 64;

  {  // stage Q with affine
    int r = tid >> 1, seg = (tid & 1) * 32;
    const float* g = Qsrc + boff + (long)(q0 + r) * 64 + seg;
#pragma unroll
    for (int e = 0; e < 32; e += 4) {
      float4 v = *(const float4*)(g + e);
      Qs[r][seg + e + 0] = (bf16_t)(aq[seg + e + 0] * v.x + cq[seg + e + 0]);
      Qs[r][seg + e + 1] = (bf16_t)(aq[seg + e + 1] * v.y + cq[seg + e + 1]);
      Qs[r][seg + e + 2] = (bf16_t)(aq[seg + e + 2] * v.z + cq[seg + e + 2]);
      Qs[r][seg + e + 3] = (bf16_t)(aq[seg + e + 3] * v.w + cq[seg + e + 3]);
    }
  }
  __syncthreads();
  v16bf qA0 = lds_load_a_frag(&Qs[wv * 16][0], 72, lane);
  v16bf qA1 = lds_load_a_frag(&Qs[wv * 16][32], 72, lane);
  __syncthreads();

  float rmax[8], lsum[8];
  v8f O[4];
#pragma unroll
  for (int j = 0; j < 8; ++j) { rmax[j] = -3.0e38f; lsum[j] = 0.0f; }
#pragma unroll
  for (int t = 0; t < 4; ++t)
#pragma unroll
    for (int e = 0; e < 8; ++e) O[t][e] = 0.0f;

  int col = lane & 15, half = (lane >> 4) * 8;

  for (int kt = 0; kt < Nn; kt += 64) {
    {  // stage K (row-major) and V (transposed) with affine
      int r = tid >> 2, seg = (tid & 3) * 16;
      const float* gk = Ksrc + boff + (long)(kt + r) * 64 + seg;
      const float* gv = Vsrc + boff + (long)(kt + r) * 64 + seg;
      if (kt + 64 < Nn) {  // prefetch next tiles (global_prefetch_b8)
        __builtin_prefetch(gk + 64 * 64, 0, 0);
        __builtin_prefetch(gv + 64 * 64, 0, 0);
      }
#pragma unroll
      for (int e = 0; e < 16; e += 4) {
        float4 vk = *(const float4*)(gk + e);
        float4 vv = *(const float4*)(gv + e);
        int m = seg + e;
        Kt[r][m + 0] = (bf16_t)(ak[m + 0] * vk.x + ck[m + 0]);
        Kt[r][m + 1] = (bf16_t)(ak[m + 1] * vk.y + ck[m + 1]);
        Kt[r][m + 2] = (bf16_t)(ak[m + 2] * vk.z + ck[m + 2]);
        Kt[r][m + 3] = (bf16_t)(ak[m + 3] * vk.w + ck[m + 3]);
        Vt[m + 0][r] = (bf16_t)(av[m + 0] * vv.x + cv[m + 0]);
        Vt[m + 1][r] = (bf16_t)(av[m + 1] * vv.y + cv[m + 1]);
        Vt[m + 2][r] = (bf16_t)(av[m + 2] * vv.z + cv[m + 2]);
        Vt[m + 3][r] = (bf16_t)(av[m + 3] * vv.w + cv[m + 3]);
      }
    }
    __syncthreads();

    // S = Q * K^T : 16 q-rows x 64 keys per wave
    v8f S[4];
#pragma unroll
    for (int jb = 0; jb < 4; ++jb) {
      v16bf b0 = lds_load_b_frag(&Kt[jb * 16][0], 72, lane);
      v16bf b1 = lds_load_b_frag(&Kt[jb * 16][32], 72, lane);
      v8f s;
#pragma unroll
      for (int e = 0; e < 8; ++e) s[e] = 0.0f;
      s = wmma_bf16(qA0, b0, s);
      s = wmma_bf16(qA1, b1, s);
      S[jb] = s;
    }

    // online softmax
#pragma unroll
    for (int j = 0; j < 8; ++j) {
      float mc = fmaxf(fmaxf(S[0][j], S[1][j]), fmaxf(S[2][j], S[3][j]));
      mc = red16max(mc);
      float mn = fmaxf(rmax[j], mc);
      float corr = __expf(rmax[j] - mn);
      rmax[j] = mn;
      float rs = 0.0f;
#pragma unroll
      for (int f = 0; f < 4; ++f) {
        float p = __expf(S[f][j] - mn);
        S[f][j] = p;
        rs += p;
      }
      rs = red16sum(rs);
      lsum[j] = lsum[j] * corr + rs;
#pragma unroll
      for (int f = 0; f < 4; ++f) O[f][j] *= corr;
    }

    // P (C-layout) -> LDS (wave-private rows of Qs) -> A-layout fragments
#pragma unroll
    for (int f = 0; f < 4; ++f)
#pragma unroll
      for (int j = 0; j < 8; ++j)
        Qs[wv * 16 + j + half][f * 16 + col] = (bf16_t)S[f][j];
    asm volatile("s_wait_dscnt 0" ::: "memory");  // same-wave LDS RAW
    v16bf pA0 = lds_load_a_frag(&Qs[wv * 16][0], 72, lane);
    v16bf pA1 = lds_load_a_frag(&Qs[wv * 16][32], 72, lane);

    // O += P * V
#pragma unroll
    for (int mb = 0; mb < 4; ++mb) {
      v16bf b0 = lds_load_b_frag(&Vt[mb * 16][0], 72, lane);
      v16bf b1 = lds_load_b_frag(&Vt[mb * 16][32], 72, lane);
      O[mb] = wmma_bf16(pA0, b0, O[mb]);
      O[mb] = wmma_bf16(pA1, b1, O[mb]);
    }
    __syncthreads();
  }

  // normalize and store bf16 [b][n][64]
#pragma unroll
  for (int mb = 0; mb < 4; ++mb)
#pragma unroll
    for (int j = 0; j < 8; ++j) {
      long row = q0 + wv * 16 + j + half;
      Fout[boff + row * 64 + mb * 16 + col] = (bf16_t)(O[mb][j] / lsum[j]);
    }
}

// out = x + a[ch]*U + c[ch], ch = (i>>12)&511
__global__ void k_out(const float* x, const float* U, const float* a,
                      const float* c, float* out, long total) {
  long i = ((long)blockIdx.x * blockDim.x + threadIdx.x) * 4;
  if (i >= total) return;
  int ch = (int)((i >> 12) & 511);
  float4 xv = *(const float4*)(x + i);
  float4 uv = *(const float4*)(U + i);
  float A = a[ch], C = c[ch];
  float4 o;
  o.x = xv.x + A * uv.x + C;
  o.y = xv.y + A * uv.y + C;
  o.z = xv.z + A * uv.z + C;
  o.w = xv.w + A * uv.w + C;
  *(float4*)(out + i) = o;
}

// ---------------------------------------------------------------------------
extern "C" void kernel_launch(void* const* d_in, const int* in_sizes, int n_in,
                              void* d_out, int out_size, void* d_ws,
                              size_t ws_size, hipStream_t stream) {
  (void)in_sizes; (void)n_in; (void)out_size; (void)ws_size;
  const float* x = (const float*)d_in[0];
  const float* y = (const float*)d_in[1];
  const float* ws1 = (const float*)d_in[2];
  const float* gs1 = (const float*)d_in[3];
  const float* bs1 = (const float*)d_in[4];
  const float* ws2 = (const float*)d_in[5];
  const float* gs2 = (const float*)d_in[6];
  const float* bs2 = (const float*)d_in[7];
  const float* wx1 = (const float*)d_in[8];
  const float* gx1 = (const float*)d_in[9];
  const float* bx1 = (const float*)d_in[10];
  const float* wx2 = (const float*)d_in[11];
  const float* gx2 = (const float*)d_in[12];
  const float* bx2 = (const float*)d_in[13];
  const float* wy1 = (const float*)d_in[14];
  const float* gy1 = (const float*)d_in[15];
  const float* by1 = (const float*)d_in[16];
  const float* wy2 = (const float*)d_in[17];
  const float* gy2 = (const float*)d_in[18];
  const float* by2 = (const float*)d_in[19];
  const float* wu = (const float*)d_in[20];
  const float* gu = (const float*)d_in[21];
  const float* bu = (const float*)d_in[22];
  float* out = (float*)d_out;

  const int B = 4, Cx = 512, Cy = 256, M = 64, N = 4096;
  char* ws = (char*)d_ws;

  // workspace layout (bytes)
  const size_t o_xT = 0;                              // 16 MiB  B*N*Cx bf16
  const size_t o_yT = o_xT + (size_t)B * N * Cx * 2;  //  8 MiB  B*N*Cy bf16
  const size_t o_z1s = o_yT + (size_t)B * N * Cy * 2;
  const size_t o_z1x = o_z1s + (size_t)B * N * M * 4;
  const size_t o_z1y = o_z1x + (size_t)B * N * M * 4;
  const size_t o_z2s = o_z1y + (size_t)B * N * M * 4;
  const size_t o_z2x = o_z2s + (size_t)B * N * M * 4;
  const size_t o_z2y = o_z2x + (size_t)B * N * M * 4;
  const size_t o_fout = o_z2y + (size_t)B * N * M * 4;
  const size_t o_w = o_fout + (size_t)B * N * M * 2;
  const size_t o_stat = o_w + 253952;
  const size_t o_ac = o_stat + 1792 * 4;
  const size_t o_U = 0;  // aliases xT/yT/z1s/z1x (dead by GEMM3 time)

  bf16_t* xT = (bf16_t*)(ws + o_xT);
  bf16_t* yT = (bf16_t*)(ws + o_yT);
  float* z1s = (float*)(ws + o_z1s);
  float* z1x = (float*)(ws + o_z1x);
  float* z1y = (float*)(ws + o_z1y);
  float* z2s = (float*)(ws + o_z2s);
  float* z2x = (float*)(ws + o_z2x);
  float* z2y = (float*)(ws + o_z2y);
  bf16_t* fout = (bf16_t*)(ws + o_fout);
  bf16_t* wB = (bf16_t*)(ws + o_w);
  float* st = (float*)(ws + o_stat);
  float* ac = (float*)(ws + o_ac);
  float* U = (float*)(ws + o_U);

  // bf16 weight slots
  bf16_t* ws1b = wB;                 // 64*512
  bf16_t* wx1b = ws1b + 64 * 512;    // 64*512
  bf16_t* wy1b = wx1b + 64 * 512;    // 64*256
  bf16_t* ws2b = wy1b + 64 * 256;    // 64*64
  bf16_t* wx2b = ws2b + 64 * 64;
  bf16_t* wy2b = wx2b + 64 * 64;
  bf16_t* wub = wy2b + 64 * 64;      // 512*64

  // stats slots
  float *sum_s1 = st + 0, *sq_s1 = st + 64;
  float *sum_x1 = st + 128, *sq_x1 = st + 192;
  float *sum_y1 = st + 256, *sq_y1 = st + 320;
  float *sum_s2 = st + 384, *sq_s2 = st + 448;
  float *sum_x2 = st + 512, *sq_x2 = st + 576;
  float *sum_y2 = st + 640, *sq_y2 = st + 704;
  float *sum_u = st + 768, *sq_u = st + 1280;
  float *a_s1 = ac + 0, *c_s1 = ac + 64;
  float *a_x1 = ac + 128, *c_x1 = ac + 192;
  float *a_y1 = ac + 256, *c_y1 = ac + 320;
  float *a_s2 = ac + 384, *c_s2 = ac + 448;
  float *a_x2 = ac + 512, *c_x2 = ac + 576;
  float *a_y2 = ac + 640, *c_y2 = ac + 704;
  float *a_u = ac + 768, *c_u = ac + 1280;

  const float invPN = 1.0f / (float)(B * N);

  // 0) zero stat accumulators (deterministic each call)
  k_zero<<<7, 256, 0, stream>>>(st, 1792);

  // 1) weights -> bf16
  k_cvt_bf16<<<(64 * 512 + 255) / 256, 256, 0, stream>>>(ws1, ws1b, 64 * 512);
  k_cvt_bf16<<<(64 * 512 + 255) / 256, 256, 0, stream>>>(wx1, wx1b, 64 * 512);
  k_cvt_bf16<<<(64 * 256 + 255) / 256, 256, 0, stream>>>(wy1, wy1b, 64 * 256);
  k_cvt_bf16<<<(64 * 64 + 255) / 256, 256, 0, stream>>>(ws2, ws2b, 64 * 64);
  k_cvt_bf16<<<(64 * 64 + 255) / 256, 256, 0, stream>>>(wx2, wx2b, 64 * 64);
  k_cvt_bf16<<<(64 * 64 + 255) / 256, 256, 0, stream>>>(wy2, wy2b, 64 * 64);
  k_cvt_bf16<<<(512 * 64 + 255) / 256, 256, 0, stream>>>(wu, wub, 512 * 64);

  // 2) activations -> [N,C] bf16 (transpose)
  k_transpose_cvt<<<dim3(N / 32, Cx / 32, B), dim3(32, 8), 0, stream>>>(x, xT, Cx, N);
  k_transpose_cvt<<<dim3(N / 32, Cy / 32, B), dim3(32, 8), 0, stream>>>(y, yT, Cy, N);

  // 3) stage-1 GEMMs: z1 = Act @ W1^T  (I=N, J=64)
  dim3 g1(N / 128, 1, B);
  k_gemm_nt<false><<<g1, 256, 0, stream>>>(xT, nullptr, nullptr, ws1b, z1s,
                                           N, 64, Cx, (long)N * Cx, 0, (long)N * 64);
  k_gemm_nt<false><<<g1, 256, 0, stream>>>(xT, nullptr, nullptr, wx1b, z1x,
                                           N, 64, Cx, (long)N * Cx, 0, (long)N * 64);
  k_gemm_nt<false><<<g1, 256, 0, stream>>>(yT, nullptr, nullptr, wy1b, z1y,
                                           N, 64, Cy, (long)N * Cy, 0, (long)N * 64);

  long tz = (long)B * N * 64;
  k_stats_cols<<<1024, 256, 0, stream>>>(z1s, tz, sum_s1, sq_s1);
  k_stats_cols<<<1024, 256, 0, stream>>>(z1x, tz, sum_x1, sq_x1);
  k_stats_cols<<<1024, 256, 0, stream>>>(z1y, tz, sum_y1, sq_y1);
  k_finalize<<<1, 64, 0, stream>>>(sum_s1, sq_s1, gs1, bs1, a_s1, c_s1, 64, invPN);
  k_finalize<<<1, 64, 0, stream>>>(sum_x1, sq_x1, gx1, bx1, a_x1, c_x1, 64, invPN);
  k_finalize<<<1, 64, 0, stream>>>(sum_y1, sq_y1, gy1, by1, a_y1, c_y1, 64, invPN);

  // 4) stage-2 GEMMs with fused BN1 affine on A
  k_gemm_nt<true><<<g1, 256, 0, stream>>>(z1s, a_s1, c_s1, ws2b, z2s,
                                          N, 64, 64, (long)N * 64, 0, (long)N * 64);
  k_gemm_nt<true><<<g1, 256, 0, stream>>>(z1x, a_x1, c_x1, wx2b, z2x,
                                          N, 64, 64, (long)N * 64, 0, (long)N * 64);
  k_gemm_nt<true><<<g1, 256, 0, stream>>>(z1y, a_y1, c_y1, wy2b, z2y,
                                          N, 64, 64, (long)N * 64, 0, (long)N * 64);

  k_stats_cols<<<1024, 256, 0, stream>>>(z2s, tz, sum_s2, sq_s2);
  k_stats_cols<<<1024, 256, 0, stream>>>(z2x, tz, sum_x2, sq_x2);
  k_stats_cols<<<1024, 256, 0, stream>>>(z2y, tz, sum_y2, sq_y2);
  k_finalize<<<1, 64, 0, stream>>>(sum_s2, sq_s2, gs2, bs2, a_s2, c_s2, 64, invPN);
  k_finalize<<<1, 64, 0, stream>>>(sum_x2, sq_x2, gx2, bx2, a_x2, c_x2, 64, invPN);
  k_finalize<<<1, 64, 0, stream>>>(sum_y2, sq_y2, gy2, by2, a_y2, c_y2, 64, invPN);

  // 5) flash attention: fout = softmax(fx fy^T) fself  (BN2 affines fused)
  k_attn<<<dim3(N / 128, B), 256, 0, stream>>>(z2x, a_x2, c_x2, z2y, a_y2, c_y2,
                                               z2s, a_s2, c_s2, fout, N);

  // 6) up-projection: U[b][co][n] = sum_m wu[co][m] * fout[b][n][m]
  dim3 g3(Cx / 128, N / 64, B);
  k_gemm_nt<false><<<g3, 256, 0, stream>>>(wub, nullptr, nullptr, fout, U,
                                           Cx, N, 64, 0, (long)N * 64,
                                           (long)Cx * N);

  // 7) final BN stats + residual epilogue
  k_stats_rows<<<B * Cx, 256, 0, stream>>>(U, N, Cx - 1, sum_u, sq_u);
  k_finalize<<<2, 256, 0, stream>>>(sum_u, sq_u, gu, bu, a_u, c_u, Cx, invPN);
  long total = (long)B * Cx * N;
  k_out<<<(unsigned)(total / 4 / 256), 256, 0, stream>>>(x, U, a_u, c_u, out, total);
}
